// MRS_36721970381386
// MI455X (gfx1250) — matrix operand order repaired and checked by reading, not compiled, not run on hardware
//
#include <hip/hip_runtime.h>

// ---------------------------------------------------------------------------
// Types for CDNA5 WMMA (wave32): V_WMMA_F32_16X16X32_BF16
// ---------------------------------------------------------------------------
typedef __attribute__((ext_vector_type(16))) __bf16         v16bf;
typedef __attribute__((ext_vector_type(2)))  __bf16         v2bf;
typedef __attribute__((ext_vector_type(16))) unsigned short v16u;
typedef __attribute__((ext_vector_type(8)))  unsigned int   v8u;
typedef __attribute__((ext_vector_type(8)))  float          v8f;

#if defined(__gfx1250__) && __has_builtin(__builtin_amdgcn_tensor_load_to_lds) && \
    __has_builtin(__builtin_amdgcn_s_wait_tensorcnt)
#define USE_TDM 1
#else
#define USE_TDM 0
#endif

__device__ __forceinline__ unsigned short f2bf(float f) {
  unsigned int u = __float_as_uint(f);
  // round-to-nearest-even f32 -> bf16
  return (unsigned short)((u + 0x7FFFu + ((u >> 16) & 1u)) >> 16);
}

// packed f32x2 -> bf16x2 in one 32-bit reg (v_cvt_pk_bf16_f32 when available)
#if defined(__gfx1250__) && __has_builtin(__builtin_amdgcn_cvt_pk_bf16_f32)
__device__ __forceinline__ unsigned int pk2(float x, float y) {
  v2bf r = __builtin_amdgcn_cvt_pk_bf16_f32(x, y);
  return __builtin_bit_cast(unsigned int, r);
}
#else
__device__ __forceinline__ unsigned int pk2(float x, float y) {
  return (unsigned int)f2bf(x) | ((unsigned int)f2bf(y) << 16);
}
#endif

// ---------------------------------------------------------------------------
// Pack B (K x 64, f32 row-major) into WMMA B-fragment order (bf16).
// B frag (32x16): lane = n + 16*s holds column n, half j = B[kb*32+8*s+(j&7)+16*(j>>3), ct*16+n]
// dst index = (((kb*4 + ct)*32 + lane)*16 + j)
// ---------------------------------------------------------------------------
__global__ void pack_b_kernel(const float* __restrict__ src,
                              unsigned short* __restrict__ dst, int K) {
  int idx = blockIdx.x * blockDim.x + threadIdx.x;
  if (idx >= K * 64) return;
  int j    = idx & 15;
  int lane = (idx >> 4) & 31;
  int blk  = idx >> 9;          // kb*4 + ct
  int ct   = blk & 3;
  int kb   = blk >> 2;
  int n = lane & 15, s = lane >> 4;
  int k = kb * 32 + s * 8 + (j & 7) + ((j >> 3) << 4);
  dst[idx] = f2bf(src[(size_t)k * 64 + (size_t)(ct * 16 + n)]);
}

#if USE_TDM
// Issue one TDM transfer: 4KB contiguous (1024 x 4B elements) global -> LDS.
// D# built per CDNA5 ISA 8.3/8.4 (count=1, type=2, data_size=4B, 1-D tile).
__device__ __forceinline__ void tdm_load_4kb(unsigned lds_off, const unsigned short* gptr) {
  typedef __attribute__((ext_vector_type(4))) unsigned int u32x4;
  typedef __attribute__((ext_vector_type(8))) int          i32x8;
  typedef __attribute__((ext_vector_type(4))) int          i32x4;
  unsigned long long ga = (unsigned long long)(uintptr_t)gptr;
  u32x4 g0;
  g0[0] = 1u;                                   // count=1 (valid), load, user mode
  g0[1] = lds_off;                              // lds_addr (bytes)
  g0[2] = (unsigned int)(ga & 0xFFFFFFFFull);   // global_addr[31:0]
  g0[3] = (unsigned int)((ga >> 32) & 0x01FFFFFFull) | 0x80000000u; // addr[56:32] | type=2
  i32x8 g1;
  g1[0] = 0x00020000;            // data_size=2 (4 bytes), workgroup_mask=0
  g1[1] = (int)(1024u << 16);    // tensor_dim0 = 1024 (low 16 bits -> bits 63:48)
  g1[2] = (int)(1u << 16);       // tensor_dim0 hi=0 | tensor_dim1 = 1
  g1[3] = (int)(1024u << 16);    // tensor_dim1 hi=0 | tile_dim0 = 1024
  g1[4] = 1;                     // tile_dim1 = 1, tile_dim2 = 0
  g1[5] = 1024;                  // tensor_dim0_stride[31:0]
  g1[6] = 0;                     // stride hi | tensor_dim1_stride lo
  g1[7] = 0;
  i32x4 z4 = {0, 0, 0, 0};
  i32x8 z8 = {0, 0, 0, 0, 0, 0, 0, 0};
  // 6-arg flavor (therock/clang-23 headers): groups 2/3/aux unused for 2D tensors
  __builtin_amdgcn_tensor_load_to_lds(g0, g1, z4, z4, z8, 0);
}
#endif

// ---------------------------------------------------------------------------
// C[M x 64] = act( A[M x K](f32 -> bf16 on load) @ Bp + bias )
// One wave -> 16x64 tile (4x v_wmma_f32_16x16x32_bf16 per 32-wide K-step).
// B K-chunks (4KB) staged in LDS via TDM (double-buffered) when available.
// act: 0 = none, 1 = leaky_relu(0.01)
// ---------------------------------------------------------------------------
__global__ __launch_bounds__(256)
void gemm_bf16_n64(const float* __restrict__ A,
                   const unsigned short* __restrict__ Bp,
                   float* __restrict__ C, int M, int K,
                   const float* __restrict__ bias, int act) {
  const int lane = threadIdx.x & 31;
  const int wave = threadIdx.x >> 5;
  int row0 = (blockIdx.x * 8 + wave) * 16;
  if (row0 >= M) row0 = M - 16;   // clamp (keeps barriers uniform; M%128==0 here)

  const int mrow = lane & 15;
  const int s    = lane >> 4;
  const float* arow = A + (size_t)(row0 + mrow) * (size_t)K + (size_t)(8 * s);

  v8f acc0 = {}, acc1 = {}, acc2 = {}, acc3 = {};
  const int nk = K >> 5;

#if USE_TDM
  __shared__ __align__(64) unsigned short ldsB[2][2048];   // 2 x 4KB double buffer
  const unsigned lds_base =
      (unsigned)(unsigned long long)(__attribute__((address_space(3))) unsigned short*)&ldsB[0][0];
  if (wave == 0) tdm_load_4kb(lds_base, Bp);               // prefetch chunk 0
  int cur = 0;
#endif

  for (int kb = 0; kb < nk; ++kb) {
    const int k0 = kb * 32;

    // ---- A fragment: lane(m,s) holds A[m, k0+8s+0..7] and A[m, k0+8s+16..23]
    const float4 a0 = *(const float4*)(arow + k0);
    const float4 a1 = *(const float4*)(arow + k0 + 4);
    const float4 a2 = *(const float4*)(arow + k0 + 16);
    const float4 a3 = *(const float4*)(arow + k0 + 20);
    if (k0 + 256 < K) __builtin_prefetch(arow + k0 + 256, 0, 0);
    v8u au;
    au[0] = pk2(a0.x, a0.y); au[1] = pk2(a0.z, a0.w);
    au[2] = pk2(a1.x, a1.y); au[3] = pk2(a1.z, a1.w);
    au[4] = pk2(a2.x, a2.y); au[5] = pk2(a2.z, a2.w);
    au[6] = pk2(a3.x, a3.y); au[7] = pk2(a3.z, a3.w);
    const v16bf a = __builtin_bit_cast(v16bf, au);

#if USE_TDM
    if (wave == 0) __builtin_amdgcn_s_wait_tensorcnt(0);   // chunk kb landed in LDS
    __syncthreads();                                       // ...visible to all waves
    if (wave == 0 && kb + 1 < nk)                          // prefetch chunk kb+1
      tdm_load_4kb(lds_base + (unsigned)((cur ^ 1) * 4096),
                   Bp + (size_t)(kb + 1) * 2048);
    const unsigned short* bln = &ldsB[cur][lane * 16];
    const v16bf b0 = __builtin_bit_cast(v16bf, *(const v16u*)(bln));
    const v16bf b1 = __builtin_bit_cast(v16bf, *(const v16u*)(bln + 512));
    const v16bf b2 = __builtin_bit_cast(v16bf, *(const v16u*)(bln + 1024));
    const v16bf b3 = __builtin_bit_cast(v16bf, *(const v16u*)(bln + 1536));
    cur ^= 1;
#else
    const unsigned short* bln = Bp + (size_t)kb * 2048 + (size_t)lane * 16;
    const v16bf b0 = __builtin_bit_cast(v16bf, *(const v16u*)(bln));
    const v16bf b1 = __builtin_bit_cast(v16bf, *(const v16u*)(bln + 512));
    const v16bf b2 = __builtin_bit_cast(v16bf, *(const v16u*)(bln + 1024));
    const v16bf b3 = __builtin_bit_cast(v16bf, *(const v16u*)(bln + 1536));
#endif

    acc0 = __builtin_amdgcn_wmma_f32_16x16x32_bf16(false, a, false, b0, (short)0, acc0, false, false);
    acc1 = __builtin_amdgcn_wmma_f32_16x16x32_bf16(false, a, false, b1, (short)0, acc1, false, false);
    acc2 = __builtin_amdgcn_wmma_f32_16x16x32_bf16(false, a, false, b2, (short)0, acc2, false, false);
    acc3 = __builtin_amdgcn_wmma_f32_16x16x32_bf16(false, a, false, b3, (short)0, acc3, false, false);
  }

  // ---- epilogue: C element r of lane -> row = r + 8*s, col = ct*16 + (lane&15)
  const int ncol = lane & 15;
  float bb0 = 0.f, bb1 = 0.f, bb2 = 0.f, bb3 = 0.f;
  if (bias) {
    bb0 = bias[ncol]; bb1 = bias[16 + ncol]; bb2 = bias[32 + ncol]; bb3 = bias[48 + ncol];
  }
#pragma unroll
  for (int r = 0; r < 8; ++r) {
    const int row = row0 + r + 8 * s;
    float* crow = C + (size_t)row * 64 + ncol;
    float v0 = acc0[r] + bb0, v1 = acc1[r] + bb1, v2 = acc2[r] + bb2, v3 = acc3[r] + bb3;
    if (act == 1) {
      v0 = v0 > 0.f ? v0 : 0.01f * v0;
      v1 = v1 > 0.f ? v1 : 0.01f * v1;
      v2 = v2 > 0.f ? v2 : 0.01f * v2;
      v3 = v3 > 0.f ? v3 : 0.01f * v3;
    }
    crow[0] = v0; crow[16] = v1; crow[32] = v2; crow[48] = v3;
  }
}

// ---------------------------------------------------------------------------
// Elementwise helpers
// ---------------------------------------------------------------------------
__global__ void mean2_kernel(const float* __restrict__ a, const float* __restrict__ b,
                             float* __restrict__ o, int n) {
  int i = blockIdx.x * blockDim.x + threadIdx.x;
  if (i < n) o[i] = 0.5f * (a[i] + b[i]);
}

// W_sum[k,n] = sum_h w_cat[h*64 + k, n]
__global__ void wsum_kernel(const float* __restrict__ wcat, float* __restrict__ wsum) {
  int i = blockIdx.x * blockDim.x + threadIdx.x;
  if (i < 64 * 64) {
    int k = i >> 6, n = i & 63;
    wsum[i] = wcat[k * 64 + n] + wcat[(64 + k) * 64 + n] +
              wcat[(128 + k) * 64 + n] + wcat[(192 + k) * 64 + n];
  }
}

// o[r,:] = base[r,:] + rate * src[r,:] / max(||src[r,:]||, 1e-12)
__global__ void addnorm_kernel(const float* __restrict__ base, const float* __restrict__ src,
                               float* __restrict__ o, float rate, int rows) {
  int r = blockIdx.x * blockDim.x + threadIdx.x;
  if (r >= rows) return;
  const float* sp = src + (size_t)r * 64;
  const float* bp = base + (size_t)r * 64;
  float* op = o + (size_t)r * 64;
  float ss = 0.f;
  for (int j = 0; j < 64; ++j) ss += sp[j] * sp[j];
  float inv = rate / fmaxf(sqrtf(ss), 1e-12f);
  for (int j = 0; j < 64; ++j) op[j] = bp[j] + inv * sp[j];
}

// in-place row softmax over 64
__global__ void softmax64_kernel(float* __restrict__ x, int rows) {
  int r = blockIdx.x * blockDim.x + threadIdx.x;
  if (r >= rows) return;
  float* p = x + (size_t)r * 64;
  float mx = -3.402823e38f;
  for (int j = 0; j < 64; ++j) mx = fmaxf(mx, p[j]);
  float sum = 0.f;
  for (int j = 0; j < 64; ++j) sum += expf(p[j] - mx);
  float inv = 1.f / sum;
  for (int j = 0; j < 64; ++j) p[j] = expf(p[j] - mx) * inv;
}

// out = (g0+g1+g2)/3 + 0.02*norm(f0 row) + 0.02*norm(f1 row); write to two dests
__global__ void final_kernel(const float* __restrict__ g0, const float* __restrict__ g1,
                             const float* __restrict__ g2, const float* __restrict__ f0,
                             const float* __restrict__ f1, float* __restrict__ out1,
                             float* __restrict__ out2, int rows) {
  int r = blockIdx.x * blockDim.x + threadIdx.x;
  if (r >= rows) return;
  const size_t off = (size_t)r * 64;
  const float *p0 = g0 + off, *p1 = g1 + off, *p2 = g2 + off;
  const float *q0 = f0 + off, *q1 = f1 + off;
  float n0 = 0.f, n1 = 0.f;
  for (int j = 0; j < 64; ++j) { n0 += q0[j] * q0[j]; n1 += q1[j] * q1[j]; }
  float i0 = 0.02f / fmaxf(sqrtf(n0), 1e-12f);
  float i1 = 0.02f / fmaxf(sqrtf(n1), 1e-12f);
  for (int j = 0; j < 64; ++j) {
    float v = (p0[j] + p1[j] + p2[j]) * (1.f / 3.f) + i0 * q0[j] + i1 * q1[j];
    out1[off + j] = v;
    out2[off + j] = v;
  }
}

// ---------------------------------------------------------------------------
// Orchestration
// ---------------------------------------------------------------------------
extern "C" void kernel_launch(void* const* d_in, const int* in_sizes, int n_in,
                              void* d_out, int out_size, void* d_ws, size_t ws_size,
                              hipStream_t stream) {
  (void)in_sizes; (void)n_in; (void)out_size; (void)ws_size;
  const int U = 8192, I = 6144, D = 64;

  const float* ui     = (const float*)d_in[0];
  const float* iu     = (const float*)d_in[1];
  const float* mmui   = (const float*)d_in[2];
  const float* mmiu   = (const float*)d_in[3];
  const float* featsA[2] = { (const float*)d_in[4], (const float*)d_in[5] };
  const int    Kf[2]     = { 4096, 384 };
  const float* w1a[2] = { (const float*)d_in[6],  (const float*)d_in[10] };
  const float* b1a[2] = { (const float*)d_in[7],  (const float*)d_in[11] };
  const float* w2a[2] = { (const float*)d_in[8],  (const float*)d_in[12] };
  const float* b2a[2] = { (const float*)d_in[9],  (const float*)d_in[13] };
  const float* uemb   = (const float*)d_in[14];
  const float* iemb   = (const float*)d_in[15];
  const float* wcat   = (const float*)d_in[17];   // w_q (d_in[16]) is numerically dead

  // output slots (reference tuple order, flattened)
  float* out    = (float*)d_out;
  float* o_ug   = out;
  float* o_ig   = o_ug  + (size_t)U * D;
  float* o_itf  = o_ig  + (size_t)I * D;          // [2][I*D] item_feats
  float* o_usf  = o_itf + (size_t)2 * I * D;      // [2][U*D] user_feats
  float* o_ug2  = o_usf + (size_t)2 * U * D;
  float* o_ig2  = o_ug2 + (size_t)U * D;
  float* o_uids = o_ig2 + (size_t)I * D;          // [2][U*D]
  float* o_iids = o_uids + (size_t)2 * U * D;     // [2][I*D]

  // workspace carve-out
  char* wptr = (char*)d_ws;
  auto alloc = [&](size_t bytes) -> void* {
    void* p = (void*)wptr;
    wptr += (bytes + 255) & ~(size_t)255;
    return p;
  };
  float* W0   = (float*)alloc((size_t)U * D * 4);
  float* W1   = (float*)alloc((size_t)U * D * 4);
  float* W2   = (float*)alloc((size_t)U * D * 4);
  float* Gu0  = (float*)alloc((size_t)U * D * 4);
  float* Gu1  = (float*)alloc((size_t)U * D * 4);
  float* Gi0  = (float*)alloc((size_t)I * D * 4);
  float* Gi1  = (float*)alloc((size_t)I * D * 4);
  float* UE   = (float*)alloc((size_t)U * D * 4);
  float* IE   = (float*)alloc((size_t)I * D * 4);
  float* WSum = (float*)alloc((size_t)64 * 64 * 4);
  unsigned short* P0 = (unsigned short*)alloc((size_t)U * D * 2);  // transient packs
  unsigned short* Pi = (unsigned short*)alloc((size_t)I * D * 2);  // packed item_id_emb
  unsigned short* Pu = (unsigned short*)alloc((size_t)U * D * 2);  // packed user_id_emb

  auto packto = [&](const float* src, unsigned short* dst, int K) {
    int n = K * 64;
    pack_b_kernel<<<(n + 255) / 256, 256, 0, stream>>>(src, dst, K);
  };
  auto gemm = [&](const float* A, const unsigned short* Bp, float* C, int M, int K,
                  const float* bias, int act) {
    gemm_bf16_n64<<<(M + 127) / 128, 256, 0, stream>>>(A, Bp, C, M, K, bias, act);
  };

  packto(iemb, Pi, I);
  packto(uemb, Pu, U);

  // ---- per-modality: encoder, 2-layer propagation, id-graph matmuls -------
  for (int m = 0; m < 2; ++m) {
    packto(w1a[m], P0, Kf[m]);
    gemm(featsA[m], P0, W0, I, Kf[m], b1a[m], 1);            // leaky(feats@w1+b1)
    packto(w2a[m], P0, 64);
    gemm(W0, P0, W2, I, 64, b2a[m], 1);                      // item_f (encoded)
    packto(W2, P0, I);
    gemm(ui, P0, W1, U, I, nullptr, 0);                      // user_f (layer 1)
    packto(W1, P0, U);
    gemm(iu, P0, W2, I, U, nullptr, 0);                      // item_f (layer 1)
    packto(W2, P0, I);
    gemm(ui, P0, o_usf + (size_t)m * U * D, U, I, nullptr, 0);   // user_f final
    packto(o_usf + (size_t)m * U * D, P0, U);
    gemm(iu, P0, o_itf + (size_t)m * I * D, I, U, nullptr, 0);   // item_f final
    gemm(mmui + (size_t)m * U * I, Pi, o_uids + (size_t)m * U * D, U, I, nullptr, 0);
    gemm(mmiu + (size_t)m * I * U, Pu, o_iids + (size_t)m * I * D, I, U, nullptr, 0);
  }

  // ---- MHSA collapses to (mean over M) @ W_sum ----------------------------
  mean2_kernel<<<(U * D + 255) / 256, 256, 0, stream>>>(o_uids, o_uids + (size_t)U * D, W0, U * D);
  mean2_kernel<<<(I * D + 255) / 256, 256, 0, stream>>>(o_iids, o_iids + (size_t)I * D, W1, I * D);
  wsum_kernel<<<(64 * 64 + 255) / 256, 256, 0, stream>>>(wcat, WSum);
  packto(WSum, P0, 64);
  gemm(W0, P0, UE, U, 64, nullptr, 0);                       // user_emb
  gemm(W1, P0, IE, I, 64, nullptr, 0);                       // item_emb

  addnorm_kernel<<<(U + 255) / 256, 256, 0, stream>>>(uemb, UE, Gu0, 0.36f, U);
  addnorm_kernel<<<(I + 255) / 256, 256, 0, stream>>>(iemb, IE, Gi0, 0.36f, I);

  // ---- final 2-layer ui propagation (softmax on last layer) ---------------
  packto(Gi0, P0, I);  gemm(ui, P0, Gu1, U, I, nullptr, 0);
  packto(Gu1, P0, U);  gemm(iu, P0, Gi1, I, U, nullptr, 0);
  packto(Gi1, P0, I);  gemm(ui, P0, W0, U, I, nullptr, 0);
  softmax64_kernel<<<(U + 255) / 256, 256, 0, stream>>>(W0, U);   // u_g2
  packto(W0, P0, U);   gemm(iu, P0, W1, I, U, nullptr, 0);
  softmax64_kernel<<<(I + 255) / 256, 256, 0, stream>>>(W1, I);   // i_g2

  // ---- combine: mean of 3 + 0.02 * l2norm(feats) per modality -------------
  final_kernel<<<(U + 255) / 256, 256, 0, stream>>>(
      Gu0, Gu1, W0, o_usf, o_usf + (size_t)U * D, o_ug, o_ug2, U);
  final_kernel<<<(I + 255) / 256, 256, 0, stream>>>(
      Gi0, Gi1, W1, o_itf, o_itf + (size_t)I * D, o_ig, o_ig2, I);
}
